// MosaiqQuantumGenerator_56126632624573
// MI455X (gfx1250) — matrix operand order repaired and checked by reading, not compiled, not run on hardware
//
#include <hip/hip_runtime.h>
#include <hip/hip_bf16.h>

#define NQ    10
#define NGEN  4
#define QD    6
#define BATCH 1024
#define WAVES 4   // circuits (waves) per block

typedef _Float16 v16h __attribute__((ext_vector_type(16)));
typedef _Float16 h8   __attribute__((ext_vector_type(8)));
typedef float    v8f  __attribute__((ext_vector_type(8)));

union V16 { v16h v; h8 h[2]; };

__device__ __forceinline__ v8f wmma16(v16h a, v16h b, v8f c) {
  // D(16x16,f32) = A(16x32,f16) x B(32x16,f16) + C
  return __builtin_amdgcn_wmma_f32_16x16x32_f16(false, a, false, b, (short)0, c,
                                                false, false);
}

// A-operand fragment: matrix m is 32x32 row-major f16; take rows [16*ti, 16*ti+16).
// A 16x32 layout (matches B fragment): lane L holds row M=L&15; K = 16*(L>>4)+{0..15}.
__device__ __forceinline__ v16h load_afrag(const _Float16* m, int ti, int lane) {
  int row = ti * 16 + (lane & 15);
  int k0  = (lane >> 4) * 16;
  V16 u;
  u.h[0] = *(const h8*)(m + row * 32 + k0);
  u.h[1] = *(const h8*)(m + row * 32 + k0 + 8);
  return u.v;
}

// B-operand fragment: operand N is 32x16 (cols [16*tj,16*tj+16) of a 32-wide matrix),
// stored COLUMN-major in m (m[c*32+k] = N[k][c]).
// B 32x16 layout used here (same k order as A): lane L holds col c=L&15 (+16 lanes -> K+16); VGPR v holds
// K = 16*(L>>4) + {2v, 2v+1}  -> 16 contiguous halves per lane in column-major storage.
__device__ __forceinline__ v16h load_bfrag(const _Float16* m, int tj, int lane) {
  int c  = tj * 16 + (lane & 15);
  int k0 = (lane >> 4) * 16;
  V16 u;
  u.h[0] = *(const h8*)(m + c * 32 + k0);
  u.h[1] = *(const h8*)(m + c * 32 + k0 + 8);
  return u.v;
}

__global__ __launch_bounds__(WAVES * 32)
void qgan_circuit_kernel(const float* __restrict__ noise,
                         const float* __restrict__ qp,
                         float* __restrict__ out) {
  // Per-generator layer operators (shared by all waves in the block)
  __shared__ _Float16 matA[QD][1024];   // A_l, row-major  (A-operand of stage 1)
  __shared__ _Float16 matB[QD][1024];   // B_l, row-major == B_l^T column-major (B-operand of stage 2)
  __shared__ float    trig[QD][NQ][2];  // cos/sin of w/2
  // Per-wave state: S column-major (S[c*32+r]), T row-major (T[r*32+k]); comp 0=Re,1=Im
  __shared__ _Float16 Sst[WAVES][2][1024];
  __shared__ _Float16 Tst[WAVES][2][1024];

  const int tid  = threadIdx.x;
  const int lane = tid & 31;
  const int wave = tid >> 5;
  const int g    = blockIdx.y;
  const int b    = blockIdx.x * WAVES + wave;

  // ---- cos/sin of layer weights ----
  for (int i = tid; i < QD * NQ; i += blockDim.x) {
    int l = i / NQ, q = i % NQ;
    float sn, cs;
    sincosf(0.5f * qp[(g * QD + l) * NQ + q], &sn, &cs);
    trig[l][q][0] = cs;
    trig[l][q][1] = sn;
  }
  __syncthreads();

  // ---- build A_l (wires 0..4) and B_l (wires 5..9): Kronecker of 2x2 RY ----
  // RY = [[c,-s],[s,c]] ; entry(i,j) = (i==j) ? c : (i==0 ? -s : s)
  for (int idx = tid; idx < QD * 1024; idx += blockDim.x) {
    int l = idx >> 10;
    int e = idx & 1023;
    int r = e >> 5, k = e & 31;
    float pa = 1.f, pb = 1.f;
#pragma unroll
    for (int j = 0; j < 5; ++j) {
      int rb = (r >> (4 - j)) & 1;
      int kb = (k >> (4 - j)) & 1;
      float ca = trig[l][j][0],     sa = trig[l][j][1];
      float cb = trig[l][5 + j][0], sb = trig[l][5 + j][1];
      pa *= (rb == kb) ? ca : (rb == 0 ? -sa : sa);
      pb *= (rb == kb) ? cb : (rb == 0 ? -sb : sb);
    }
    matA[l][e] = (_Float16)pa;
    matB[l][e] = (_Float16)pb;
  }
  __syncthreads();

  // ---- initial product state:  v0 = c^2 - i s^2 ,  v1 = cs - i cs  per qubit ----
  float v0r[NQ], v0i[NQ], v1r[NQ], v1i[NQ];
#pragma unroll
  for (int q = 0; q < NQ; ++q) {
    float sn, cs;
    sincosf(0.5f * noise[b * NQ + q], &sn, &cs);
    v0r[q] = cs * cs;  v0i[q] = -sn * sn;
    v1r[q] = cs * sn;  v1i[q] = -cs * sn;
  }
  _Float16* Sre = &Sst[wave][0][0];
  _Float16* Sim = &Sst[wave][1][0];
  {
    const int c = lane;                    // this lane owns column c (wires 5..9)
    float pcr = 1.f, pci = 0.f;
#pragma unroll
    for (int j = 0; j < 5; ++j) {
      int bit = (c >> (4 - j)) & 1;
      int q = 5 + j;
      float ar = bit ? v1r[q] : v0r[q];
      float ai = bit ? v1i[q] : v0i[q];
      float nr = pcr * ar - pci * ai;
      float ni = pcr * ai + pci * ar;
      pcr = nr; pci = ni;
    }
    for (int r = 0; r < 32; ++r) {
      float prr = 1.f, pri = 0.f;
#pragma unroll
      for (int j = 0; j < 5; ++j) {
        int bit = (r >> (4 - j)) & 1;
        float ar = bit ? v1r[j] : v0r[j];
        float ai = bit ? v1i[j] : v0i[j];
        float nr = prr * ar - pri * ai;
        float ni = prr * ai + pri * ar;
        prr = nr; pri = ni;
      }
      Sre[c * 32 + r] = (_Float16)(prr * pcr - pri * pci);
      Sim[c * 32 + r] = (_Float16)(prr * pci + pri * pcr);
    }
  }

  // ---- 6 layers:  S <- (A_l * S * B_l^T) .* CZmask ----
  for (int l = 0; l < QD; ++l) {
    const _Float16* Am = &matA[l][0];
    const _Float16* Bm = &matB[l][0];

    // stage 1: T = A * S    (S column-major is exactly the B-operand layout)
    v16h a0 = load_afrag(Am, 0, lane);
    v16h a1 = load_afrag(Am, 1, lane);
#pragma unroll
    for (int comp = 0; comp < 2; ++comp) {
      const _Float16* Sc = &Sst[wave][comp][0];
      _Float16* Tc = &Tst[wave][comp][0];
      v16h b0 = load_bfrag(Sc, 0, lane);
      v16h b1 = load_bfrag(Sc, 1, lane);
      const int nb = lane & 15;
      const int mb = (lane >> 4) * 8;
#pragma unroll
      for (int ti = 0; ti < 2; ++ti) {
#pragma unroll
        for (int tj = 0; tj < 2; ++tj) {
          v8f acc = {};
          acc = wmma16(ti ? a1 : a0, tj ? b1 : b0, acc);
#pragma unroll
          for (int i = 0; i < 8; ++i)          // C layout: VGPR i -> row M=mb+i, col nb
            Tc[(ti * 16 + mb + i) * 32 + tj * 16 + nb] = (_Float16)acc[i];
        }
      }
    }

    // stage 2: S = (T * B^T) .* CZ ; matB row-major == B^T column-major
    v16h bb0 = load_bfrag(Bm, 0, lane);
    v16h bb1 = load_bfrag(Bm, 1, lane);
#pragma unroll
    for (int comp = 0; comp < 2; ++comp) {
      const _Float16* Tc = &Tst[wave][comp][0];
      _Float16* Sc = &Sst[wave][comp][0];
      v16h t0 = load_afrag(Tc, 0, lane);
      v16h t1 = load_afrag(Tc, 1, lane);
      const int col0 = lane & 15;
      const int rb0  = (lane >> 4) * 8;
#pragma unroll
      for (int ti = 0; ti < 2; ++ti) {
#pragma unroll
        for (int tj = 0; tj < 2; ++tj) {
          v8f acc = {};
          acc = wmma16(ti ? t1 : t0, tj ? bb1 : bb0, acc);
          const int col   = tj * 16 + col0;
          const int rbase = ti * 16 + rb0;
          h8 pk;
#pragma unroll
          for (int i = 0; i < 8; ++i) {
            int s = ((rbase + i) << 5) | col;          // 10-bit basis index
            float sg = (__popc(s & (s >> 1)) & 1) ? -1.f : 1.f;  // CZ chain phase
            pk[i] = (_Float16)(acc[i] * sg);
          }
          *(h8*)(Sc + col * 32 + rbase) = pk;          // packed column-major store
        }
      }
    }
  }

  // ---- expectations <X_q> = sum_i Re_i*Re_flip + Im_i*Im_flip ----
  float acc[NQ];
#pragma unroll
  for (int q = 0; q < NQ; ++q) acc[q] = 0.f;
  {
    const int c = lane;
    for (int r = 0; r < 32; ++r) {
      float er = (float)Sre[c * 32 + r];
      float ei = (float)Sim[c * 32 + r];
#pragma unroll
      for (int q = 0; q < NQ; ++q) {
        const int bp = 9 - q;                 // wire q <-> bit (9-q) of s = r*32+c
        float pr, pi;
        if (bp >= 5) {
          int r2 = r ^ (1 << (bp - 5));
          pr = (float)Sre[c * 32 + r2];
          pi = (float)Sim[c * 32 + r2];
        } else {
          int c2 = c ^ (1 << bp);
          pr = (float)Sre[c2 * 32 + r];
          pi = (float)Sim[c2 * 32 + r];
        }
        acc[q] += er * pr + ei * pi;
      }
    }
  }
#pragma unroll
  for (int q = 0; q < NQ; ++q) {
    float v = acc[q];
    for (int off = 16; off > 0; off >>= 1)
      v += __shfl_xor(v, off, 32);
    if (lane == 0) out[b * (NGEN * NQ) + g * NQ + q] = v;
  }
}

extern "C" void kernel_launch(void* const* d_in, const int* in_sizes, int n_in,
                              void* d_out, int out_size, void* d_ws, size_t ws_size,
                              hipStream_t stream) {
  (void)in_sizes; (void)n_in; (void)out_size; (void)d_ws; (void)ws_size;
  const float* noise = (const float*)d_in[0];   // (BATCH, NQ) f32
  const float* qp    = (const float*)d_in[1];   // (NGEN, QD, NQ) f32
  float* out = (float*)d_out;                   // (BATCH, NGEN*NQ) f32
  dim3 grid(BATCH / WAVES, NGEN);
  qgan_circuit_kernel<<<grid, WAVES * 32, 0, stream>>>(noise, qp, out);
}
